// CGCEncoder_85186381349378
// MI455X (gfx1250) — matrix operand order repaired and checked by reading, not compiled, not run on hardware
//
#include <hip/hip_runtime.h>
#include <hip/hip_bf16.h>

typedef __bf16 bf16;
typedef __attribute__((ext_vector_type(16))) __bf16 v16bf;
typedef __attribute__((ext_vector_type(8)))  __bf16 v8bf;
typedef __attribute__((ext_vector_type(8)))  float   v8f;
typedef __attribute__((ext_vector_type(4)))  float   v4f;

namespace {
constexpr int kN    = 51200;   // nodes
constexpr int kE    = 614400;  // edges
constexpr int kOrig = 92;
constexpr int kAFL  = 64;
constexpr int kBins = 100;
constexpr int kBatch = 128;
constexpr int kAtoms = kN / kBatch;  // 400 (multiple of 16)
}

// ---------------------------------------------------------------- helpers ---

__device__ __forceinline__ v8f wmma_bf16(v16bf a, v16bf b, v8f c) {
  return __builtin_amdgcn_wmma_f32_16x16x32_bf16(
      false, a, false, b, (short)0, c, false, false);
}

__device__ __forceinline__ float leaky(float v) { return v > 0.f ? v : 0.01f * v; }

// B fragment (32x16) from transposed, pre-padded bf16 weights Bt[n][k]
// (row stride ldk, multiple of 8). Two 16B loads per lane (k 8h..8h+7, 16+8h..16+8h+7).
__device__ __forceinline__ v16bf loadB_t(const bf16* __restrict__ Bt, int ldk,
                                         int kbase, int n0, int lane) {
  int n = n0 + (lane & 15);
  int k0 = kbase + ((lane < 16) ? 0 : 8);
  const bf16* p = Bt + n * ldk + k0;
  v8bf lo = *(const v8bf*)p;
  v8bf hi = *(const v8bf*)(p + 16);
  v16bf b;
#pragma unroll
  for (int i = 0; i < 8; ++i) { b[i] = lo[i]; b[8 + i] = hi[i]; }
  return b;
}

// A fragment (16x32) from a bf16 row (this lane's M row).
__device__ __forceinline__ v16bf loadA_bf(const bf16* __restrict__ row,
                                          int kbase, int lane) {
  int kA = kbase + ((lane < 16) ? 0 : 8);
  int kB = kbase + ((lane < 16) ? 16 : 24);
  v8bf lo = *(const v8bf*)(row + kA);
  v8bf hi = *(const v8bf*)(row + kB);
  v16bf a;
#pragma unroll
  for (int i = 0; i < 8; ++i) { a[i] = lo[i]; a[8 + i] = hi[i]; }
  return a;
}

// A fragment from a f32 row (converted to bf16).
__device__ __forceinline__ v16bf loadA_f32(const float* __restrict__ row,
                                           int kbase, int lane) {
  int kA = kbase + ((lane < 16) ? 0 : 8);
  int kB = kbase + ((lane < 16) ? 16 : 24);
  v4f f0 = *(const v4f*)(row + kA);
  v4f f1 = *(const v4f*)(row + kA + 4);
  v4f f2 = *(const v4f*)(row + kB);
  v4f f3 = *(const v4f*)(row + kB + 4);
  v16bf a;
#pragma unroll
  for (int i = 0; i < 4; ++i) {
    a[i] = (bf16)f0[i]; a[4 + i] = (bf16)f1[i];
    a[8 + i] = (bf16)f2[i]; a[12 + i] = (bf16)f3[i];
  }
  return a;
}

__device__ __forceinline__ void atomAddF(float* p, float v) { unsafeAtomicAdd(p, v); }

// ------------------------------------------------------------ prep kernels ---

__global__ void k_zero(float* __restrict__ p, int n) {
  int i = blockIdx.x * blockDim.x + threadIdx.x;
  int s = gridDim.x * blockDim.x;
  for (; i < n; i += s) p[i] = 0.f;
}

// dst[n][k] = (k<K && n<N) ? src[k][n] : 0   (dst: Npad x Kpad bf16)
__global__ void k_packBt(const float* __restrict__ src, bf16* __restrict__ dst,
                         int K, int N, int Kpad, int Npad) {
  int i = blockIdx.x * blockDim.x + threadIdx.x;
  int tot = Kpad * Npad;
  int s = gridDim.x * blockDim.x;
  for (; i < tot; i += s) {
    int n = i / Kpad, k = i - n * Kpad;
    float v = (k < K && n < N) ? src[k * N + n] : 0.f;
    dst[i] = (bf16)v;
  }
}

// xa[row][k<96] = (k<92) ? x[row][k]*att[k] : 0, bf16
__global__ void k_pack_xa(const float* __restrict__ x, const float* __restrict__ att,
                          bf16* __restrict__ xa) {
  int i = blockIdx.x * blockDim.x + threadIdx.x;
  int s = gridDim.x * blockDim.x;
  for (; i < kN * 96; i += s) {
    int row = i / 96, k = i - row * 96;
    xa[i] = (bf16)((k < kOrig) ? x[row * kOrig + k] * att[k] : 0.f);
  }
}

__global__ void k_pad(const float* __restrict__ src, float* __restrict__ dst,
                      int n, int npad) {
  int i = blockIdx.x * blockDim.x + threadIdx.x;
  if (i < npad) dst[i] = (i < n) ? src[i] : 0.f;
}

// ---------------------------------------------------------------- kernels ---

// h0 = (x * attention) @ emb_W + emb_b ; writes f32 h0 and bf16 shadow hbf
__global__ void k_embed(const bf16* __restrict__ xa, const bf16* __restrict__ Wt,
                        const float* __restrict__ bias, float* __restrict__ h0,
                        bf16* __restrict__ hbf) {
  int gw = (blockIdx.x * blockDim.x + threadIdx.x) >> 5;
  int lane = threadIdx.x & 31;
  int mtile = gw >> 2;
  int nt = gw & 3;
  if (mtile >= kN / 16) return;
  int row0 = mtile * 16;
  const bf16* xr = xa + (row0 + (lane & 15)) * 96;
  v8f c = {};
#pragma unroll
  for (int kc = 0; kc < 3; ++kc) {
    v16bf a = loadA_bf(xr, kc * 32, lane);
    v16bf b = loadB_t(Wt, 96, kc * 32, nt * 16, lane);
    c = wmma_bf16(a, b, c);
  }
  int n = nt * 16 + (lane & 15);
  int mbase = (lane < 16) ? 0 : 8;
  float bb = bias[n];
#pragma unroll
  for (int i = 0; i < 8; ++i) {
    float v = c[i] + bb;
    h0[(row0 + mbase + i) * kAFL + n] = v;
    hbf[(row0 + mbase + i) * kAFL + n] = (bf16)v;
  }
}

// AutoDiscretization: e = softmax(v2@w2 + b2 + alpha*v2) @ lut, v2=leaky(t*w1+b1)
// One wave per 16-edge tile; e stored bf16 to ebuf [E,64].
__global__ void k_edge_e(const float* __restrict__ t,
                         const float* __restrict__ w1p, const float* __restrict__ b1p,
                         const bf16* __restrict__ w2t, const float* __restrict__ b2,
                         const float* __restrict__ alphaPtr,
                         const bf16* __restrict__ lutt, bf16* __restrict__ ebuf) {
  __shared__ float v3s[4][16 * 112];
  __shared__ bf16  ps[4][16 * 128];
  int w = threadIdx.x >> 5;
  int lane = threadIdx.x & 31;
  int et = (blockIdx.x * blockDim.x + threadIdx.x) >> 5;
  if (et >= kE / 16) return;
  int e0 = et * 16;
  float* v3 = v3s[w];
  bf16*  pr = ps[w];
  float alpha = alphaPtr[0];

  // A fragments for v2 (K padded to 128; w1p/b1p zero-padded => v2[k>=100]=0)
  float tm = t[e0 + (lane & 15)];
  v16bf av[4];
#pragma unroll
  for (int kc = 0; kc < 4; ++kc) {
    int kA = kc * 32 + ((lane < 16) ? 0 : 8);
    int kB = kc * 32 + ((lane < 16) ? 16 : 24);
    v4f wa0 = *(const v4f*)(w1p + kA), wa1 = *(const v4f*)(w1p + kA + 4);
    v4f ba0 = *(const v4f*)(b1p + kA), ba1 = *(const v4f*)(b1p + kA + 4);
    v4f wb0 = *(const v4f*)(w1p + kB), wb1 = *(const v4f*)(w1p + kB + 4);
    v4f bb0 = *(const v4f*)(b1p + kB), bb1 = *(const v4f*)(b1p + kB + 4);
    v16bf a;
#pragma unroll
    for (int i = 0; i < 4; ++i) {
      a[i]      = (bf16)leaky(tm * wa0[i] + ba0[i]);
      a[4 + i]  = (bf16)leaky(tm * wa1[i] + ba1[i]);
      a[8 + i]  = (bf16)leaky(tm * wb0[i] + bb0[i]);
      a[12 + i] = (bf16)leaky(tm * wb1[i] + bb1[i]);
    }
    av[kc] = a;
  }

  // v3 = v2 @ w2 (+ b2 + alpha*v2), 7 N-tiles over 100 cols
#pragma unroll 1
  for (int nt = 0; nt < 7; ++nt) {
    v8f c = {};
#pragma unroll
    for (int kc = 0; kc < 4; ++kc) {
      v16bf b = loadB_t(w2t, 128, kc * 32, nt * 16, lane);
      c = wmma_bf16(av[kc], b, c);
    }
    int n = nt * 16 + (lane & 15);
    if (n < kBins) {
      float w1n = w1p[n], b1n = b1p[n], b2n = b2[n];
      int mbase = (lane < 16) ? 0 : 8;
#pragma unroll
      for (int i = 0; i < 8; ++i) {
        float ti = t[e0 + mbase + i];
        v3[(mbase + i) * 112 + n] = c[i] + b2n + alpha * leaky(ti * w1n + b1n);
      }
    }
  }
  __syncthreads();

  // per-row softmax over 100 cols: 2 lanes per row (cols split 0..63 / 64..99),
  // cross-half combine with shfl_xor; probs bf16, zero-padded to 128.
  {
    int row = lane & 15;
    int hf = lane >> 4;                  // 0 or 1
    float* vr = v3 + row * 112;
    float mx = -1e30f;
    for (int j = 0; j < 64; ++j) {
      int jj = hf * 64 + j;
      float v = (jj < kBins) ? vr[jj] : -1e30f;
      mx = fmaxf(mx, v);
    }
    mx = fmaxf(mx, __shfl_xor(mx, 16, 32));
    float s = 0.f;
    for (int j = 0; j < 64; ++j) {
      int jj = hf * 64 + j;
      if (jj < kBins) { float p = __expf(vr[jj] - mx); vr[jj] = p; s += p; }
    }
    s += __shfl_xor(s, 16, 32);
    float inv = 1.f / s;
    bf16* prr = pr + row * 128;
    for (int j = 0; j < 64; ++j) {
      int jj = hf * 64 + j;
      prr[jj] = (bf16)((jj < kBins) ? vr[jj] * inv : 0.f);
    }
  }
  __syncthreads();

  // e = probs @ lut : [16,128p]x[128p,64]
  const bf16* prow = pr + (lane & 15) * 128;
#pragma unroll 1
  for (int nt = 0; nt < 4; ++nt) {
    v8f c = {};
#pragma unroll
    for (int kc = 0; kc < 4; ++kc) {
      v16bf a = loadA_bf(prow, kc * 32, lane);
      v16bf b = loadB_t(lutt, 128, kc * 32, nt * 16, lane);
      c = wmma_bf16(a, b, c);
    }
    int n = nt * 16 + (lane & 15);
    int mbase = (lane < 16) ? 0 : 8;
#pragma unroll
    for (int i = 0; i < 8; ++i) ebuf[(e0 + mbase + i) * kAFL + n] = (bf16)c[i];
  }
}

// Cooperative copy of the 128x192 bf16 fcW slab into LDS (49152 B).
__device__ __forceinline__ void stage_fcW(const bf16* __restrict__ fcWt, bf16* Bs) {
  const uint4* src = (const uint4*)fcWt;
  uint4* dst = (uint4*)Bs;
  for (int i = threadIdx.x; i < (128 * 192 * 2) / 16; i += blockDim.x) dst[i] = src[i];
  __syncthreads();
}

// z-tile = [h[dst]|h[src]|e] @ fcW + fcb for one 16-edge tile -> C[8] (16x128).
// A: bf16 shadow of h + bf16 e (pure b128 loads); B: fcW slab in LDS.
__device__ __forceinline__ void fc_tiles(int e0, const int* __restrict__ ei,
                                         const bf16* __restrict__ hbf,
                                         const bf16* __restrict__ ebuf,
                                         const bf16* Bs,
                                         const float* __restrict__ fcb,
                                         int lane, v8f C[8]) {
  int mm = lane & 15;
  int dst = ei[kE + e0 + mm];
  int src = ei[e0 + mm];
  const bf16* hd = hbf + dst * kAFL;
  const bf16* hs = hbf + src * kAFL;
  const bf16* er = ebuf + (e0 + mm) * kAFL;
  v16bf av[6];
  av[0] = loadA_bf(hd, 0, lane);
  av[1] = loadA_bf(hd + 32, 0, lane);
  av[2] = loadA_bf(hs, 0, lane);
  av[3] = loadA_bf(hs + 32, 0, lane);
  av[4] = loadA_bf(er, 0, lane);
  av[5] = loadA_bf(er + 32, 0, lane);
#pragma unroll 1
  for (int nt = 0; nt < 8; ++nt) {
    v8f c = {};
#pragma unroll
    for (int kc = 0; kc < 6; ++kc) {
      v16bf b = loadB_t(Bs, 192, kc * 32, nt * 16, lane);
      c = wmma_bf16(av[kc], b, c);
    }
    float bb = fcb[nt * 16 + (lane & 15)];
#pragma unroll
    for (int i = 0; i < 8; ++i) c[i] += bb;
    C[nt] = c;
  }
}

// Pass 1: per-channel sum / sumsq of z over all edges (BN1 batch stats)
__global__ void k_fc_stats(const int* __restrict__ ei, const bf16* __restrict__ hbf,
                           const bf16* __restrict__ ebuf, const bf16* __restrict__ fcWt,
                           const float* __restrict__ fcb, float* __restrict__ stats) {
  __shared__ bf16 Bs[128 * 192];
  stage_fcW(fcWt, Bs);
  int gw = (blockIdx.x * blockDim.x + threadIdx.x) >> 5;
  int nw = (gridDim.x * blockDim.x) >> 5;
  int lane = threadIdx.x & 31;
  float s[8] = {0}, q[8] = {0};
  for (int et = gw; et < kE / 16; et += nw) {
    v8f C[8];
    fc_tiles(et * 16, ei, hbf, ebuf, Bs, fcb, lane, C);
#pragma unroll
    for (int nt = 0; nt < 8; ++nt)
#pragma unroll
      for (int i = 0; i < 8; ++i) { float v = C[nt][i]; s[nt] += v; q[nt] += v * v; }
  }
#pragma unroll
  for (int nt = 0; nt < 8; ++nt) {
    int n = nt * 16 + (lane & 15);
    atomAddF(&stats[n], s[nt]);
    atomAddF(&stats[128 + n], q[nt]);
  }
}

__global__ void k_fin_bn(const float* __restrict__ stats, const float* __restrict__ g,
                         const float* __restrict__ b, float cnt, int nch,
                         float* __restrict__ scale, float* __restrict__ shift) {
  int c = blockIdx.x * blockDim.x + threadIdx.x;
  if (c < nch) {
    float mu = stats[c] / cnt;
    float var = stats[nch + c] / cnt - mu * mu;
    float sc = g[c] * rsqrtf(var + 1e-5f);
    scale[c] = sc;
    shift[c] = b[c] - mu * sc;
  }
}

// Pass 2: recompute z, apply BN1, gate, scatter-add messages into aggr
__global__ void k_fc_scatter(const int* __restrict__ ei, const bf16* __restrict__ hbf,
                             const bf16* __restrict__ ebuf, const bf16* __restrict__ fcWt,
                             const float* __restrict__ fcb, const float* __restrict__ scale,
                             const float* __restrict__ shift, float* __restrict__ aggr) {
  __shared__ bf16 Bs[128 * 192];
  stage_fcW(fcWt, Bs);
  int gw = (blockIdx.x * blockDim.x + threadIdx.x) >> 5;
  int nw = (gridDim.x * blockDim.x) >> 5;
  int lane = threadIdx.x & 31;
  float sc[8], sh[8];
#pragma unroll
  for (int nt = 0; nt < 8; ++nt) {
    int n = nt * 16 + (lane & 15);
    sc[nt] = scale[n]; sh[nt] = shift[n];
  }
  for (int et = gw; et < kE / 16; et += nw) {
    v8f C[8];
    fc_tiles(et * 16, ei, hbf, ebuf, Bs, fcb, lane, C);
    int mbase = (lane < 16) ? 0 : 8;
#pragma unroll
    for (int nt = 0; nt < 4; ++nt) {
      int n = nt * 16 + (lane & 15);
#pragma unroll
      for (int i = 0; i < 8; ++i) {
        float f = C[nt][i] * sc[nt] + sh[nt];             // nbr_filter channel n
        float g = C[nt + 4][i] * sc[nt + 4] + sh[nt + 4]; // nbr_core channel n
        float msg = fmaxf(f, 0.f) * fmaxf(g, 0.f);
        int d = ei[kE + et * 16 + mbase + i];
        atomAddF(&aggr[d * kAFL + n], msg);
      }
    }
  }
}

__global__ void k_node_stats(const float* __restrict__ aggr, float* __restrict__ stats) {
  int i = blockIdx.x * blockDim.x + threadIdx.x;
  int stride = gridDim.x * blockDim.x;   // multiple of 64
  int c = i & 63;
  float s = 0.f, q = 0.f;
  for (; i < kN * kAFL; i += stride) { float v = aggr[i]; s += v; q += v * v; }
  atomAddF(&stats[c], s);
  atomAddF(&stats[64 + c], q);
}

__global__ void k_update(const float* __restrict__ hin, const float* __restrict__ aggr,
                         const float* __restrict__ scale, const float* __restrict__ shift,
                         float* __restrict__ hout, bf16* __restrict__ hbf) {
  int i = blockIdx.x * blockDim.x + threadIdx.x;
  int stride = gridDim.x * blockDim.x;
  for (; i < kN * kAFL; i += stride) {
    int c = i & 63;
    float v = fmaxf(hin[i] + aggr[i] * scale[c] + shift[c], 0.f);
    hout[i] = v;
    hbf[i] = (bf16)v;
  }
}

// Pooling: softmax(feas @ pp_W + pp_b, axis=1), summed over (conv, atoms)
__global__ void k_pool(const float* __restrict__ feas, const bf16* __restrict__ Wt,
                       const float* __restrict__ bias, float* __restrict__ out) {
  __shared__ float Ls[4][16 * 64];
  int w = threadIdx.x >> 5;
  int lane = threadIdx.x & 31;
  int rt = (blockIdx.x * blockDim.x + threadIdx.x) >> 5;
  if (rt >= (4 * kN) / 16) return;
  int r0 = rt * 16;
  float* L = Ls[w];
  const float* fr = feas + (r0 + (lane & 15)) * kAFL;
#pragma unroll 1
  for (int nt = 0; nt < 4; ++nt) {
    v8f c = {};
#pragma unroll
    for (int kc = 0; kc < 2; ++kc) {
      v16bf a = loadA_f32(fr, kc * 32, lane);
      v16bf b = loadB_t(Wt, 64, kc * 32, nt * 16, lane);
      c = wmma_bf16(a, b, c);
    }
    int n = nt * 16 + (lane & 15);
    int mbase = (lane < 16) ? 0 : 8;
    float bb = bias[n];
#pragma unroll
    for (int i = 0; i < 8; ++i) L[(mbase + i) * 64 + n] = c[i] + bb;
  }
  __syncthreads();
  {
    int row = lane & 15;
    int hf = lane >> 4;
    float* Lr = L + row * 64;
    float mx = -1e30f;
    for (int j = 0; j < 32; ++j) mx = fmaxf(mx, Lr[hf * 32 + j]);
    mx = fmaxf(mx, __shfl_xor(mx, 16, 32));
    float s = 0.f;
    for (int j = 0; j < 32; ++j) {
      float p = __expf(Lr[hf * 32 + j] - mx);
      Lr[hf * 32 + j] = p; s += p;
    }
    s += __shfl_xor(s, 16, 32);
    float inv = 1.f / s;
    for (int j = 0; j < 32; ++j) Lr[hf * 32 + j] *= inv;
  }
  __syncthreads();
  // 16-row tile lies inside one (conv, crystal) group (400 % 16 == 0)
  int bidx = (r0 % kN) / kAtoms;
  for (int cc = lane; cc < 64; cc += 32) {
    float s = 0.f;
#pragma unroll
    for (int mm = 0; mm < 16; ++mm) s += L[mm * 64 + cc];
    atomAddF(&out[bidx * kAFL + cc], s);
  }
}

// ------------------------------------------------------------------ launch ---

extern "C" void kernel_launch(void* const* d_in, const int* in_sizes, int n_in,
                              void* d_out, int out_size, void* d_ws, size_t ws_size,
                              hipStream_t stream) {
  const float* x     = (const float*)d_in[0];
  const int*   ei    = (const int*)d_in[1];
  const float* eattr = (const float*)d_in[2];
  // d_in[3]: batchsize (=128, baked into kBatch/kAtoms)
  const float* att   = (const float*)d_in[4];
  const float* embW  = (const float*)d_in[5];
  const float* embB  = (const float*)d_in[6];
  const float* adw1  = (const float*)d_in[7];
  const float* adb1  = (const float*)d_in[8];
  const float* adw2  = (const float*)d_in[9];
  const float* adb2  = (const float*)d_in[10];
  const float* adal  = (const float*)d_in[11];
  const float* adlut = (const float*)d_in[12];
  const float* fcW   = (const float*)d_in[13];
  const float* fcb   = (const float*)d_in[14];
  const float* bn1g  = (const float*)d_in[15];
  const float* bn1b  = (const float*)d_in[16];
  const float* bn2g  = (const float*)d_in[17];
  const float* bn2b  = (const float*)d_in[18];
  const float* ppW   = (const float*)d_in[19];
  const float* ppb   = (const float*)d_in[20];
  float* out = (float*)d_out;
  (void)in_sizes; (void)n_in; (void)out_size; (void)ws_size;

  // -------- workspace layout (all 256B aligned), ~162 MB
  char* ws = (char*)d_ws;
  size_t off = 0;
  auto alloc = [&](size_t bytes) {
    void* p = ws + off;
    off += (bytes + 255) & ~(size_t)255;
    return p;
  };
  float* feas   = (float*)alloc((size_t)4 * kN * kAFL * 4);
  bf16*  ebuf   = (bf16*) alloc((size_t)kE * kAFL * 2);
  float* aggr   = (float*)alloc((size_t)kN * kAFL * 4);
  bf16*  hbf    = (bf16*) alloc((size_t)kN * kAFL * 2);
  float* stats1 = (float*)alloc(256 * 4);
  float* scale1 = (float*)alloc(128 * 4);
  float* shift1 = (float*)alloc(128 * 4);
  float* stats2 = (float*)alloc(128 * 4);
  float* scale2 = (float*)alloc(64 * 4);
  float* shift2 = (float*)alloc(64 * 4);
  bf16*  xa     = (bf16*) alloc((size_t)kN * 96 * 2);
  bf16*  embWt  = (bf16*) alloc(64 * 96 * 2);
  bf16*  ppWt   = (bf16*) alloc(64 * 64 * 2);
  bf16*  w2t    = (bf16*) alloc(3 * 112 * 128 * 2);   // per layer: 112x128
  bf16*  lutt   = (bf16*) alloc(3 * 64 * 128 * 2);    // per layer: 64x128
  bf16*  fcWt   = (bf16*) alloc(3 * 128 * 192 * 2);   // per layer: 128x192
  float* w1p    = (float*)alloc(3 * 128 * 4);
  float* b1p    = (float*)alloc(3 * 128 * 4);

  // -------- prep: pack transposed bf16 weights + padded activations
  k_pack_xa<<<4096, 256, 0, stream>>>(x, att, xa);
  k_packBt<<<24, 256, 0, stream>>>(embW, embWt, kOrig, kAFL, 96, 64);
  k_packBt<<<16, 256, 0, stream>>>(ppW, ppWt, 64, 64, 64, 64);
  for (int l = 0; l < 3; ++l) {
    k_packBt<<<56, 256, 0, stream>>>(adw2 + l * kBins * kBins, w2t + l * 112 * 128,
                                     kBins, kBins, 128, 112);
    k_packBt<<<32, 256, 0, stream>>>(adlut + l * kBins * kAFL, lutt + l * 64 * 128,
                                     kBins, kAFL, 128, 64);
    k_packBt<<<96, 256, 0, stream>>>(fcW + l * 192 * 128, fcWt + l * 128 * 192,
                                     192, 128, 192, 128);
    k_pad<<<1, 128, 0, stream>>>(adw1 + l * kBins, w1p + l * 128, kBins, 128);
    k_pad<<<1, 128, 0, stream>>>(adb1 + l * kBins, b1p + l * 128, kBins, 128);
  }

  k_zero<<<32, 256, 0, stream>>>(out, kBatch * kAFL);
  k_embed<<<(kN / 16), 128, 0, stream>>>(xa, embWt, embB, feas, hbf);

  for (int l = 0; l < 3; ++l) {
    const float* h = feas + (size_t)l * kN * kAFL;
    float* hnext = feas + (size_t)(l + 1) * kN * kAFL;

    k_zero<<<4, 256, 0, stream>>>(stats1, 256);
    k_zero<<<2, 256, 0, stream>>>(stats2, 128);
    k_zero<<<2048, 256, 0, stream>>>(aggr, kN * kAFL);

    k_edge_e<<<(kE / 16) / 4, 128, 0, stream>>>(
        eattr, w1p + l * 128, b1p + l * 128, w2t + l * 112 * 128,
        adb2 + l * kBins, adal + l, lutt + l * 64 * 128, ebuf);

    k_fc_stats<<<512, 128, 0, stream>>>(ei, hbf, ebuf, fcWt + l * 128 * 192,
                                        fcb + l * 128, stats1);
    k_fin_bn<<<1, 128, 0, stream>>>(stats1, bn1g + l * 128, bn1b + l * 128,
                                    (float)kE, 128, scale1, shift1);
    k_fc_scatter<<<960, 128, 0, stream>>>(ei, hbf, ebuf, fcWt + l * 128 * 192,
                                          fcb + l * 128, scale1, shift1, aggr);
    k_node_stats<<<256, 256, 0, stream>>>(aggr, stats2);
    k_fin_bn<<<1, 64, 0, stream>>>(stats2, bn2g + l * 64, bn2b + l * 64,
                                   (float)kN, 64, scale2, shift2);
    // note: k_update writes hbf for the NEXT layer's gathers
    k_update<<<2048, 256, 0, stream>>>(h, aggr, scale2, shift2, hnext, hbf);
  }

  k_pool<<<(4 * kN / 16) / 4, 128, 0, stream>>>(feas, ppWt, ppb, out);
}
